// AttentionCircuit_34213709480499
// MI455X (gfx1250) — matrix operand order repaired and checked by reading, not compiled, not run on hardware
//
#include <hip/hip_runtime.h>

// ---------------- problem constants ----------------
constexpr int kB  = 4;
constexpr int kS  = 1024;
constexpr int kD  = 1024;
constexpr int kK  = 16;     // top-k
constexpr int kH  = 16;     // heads
constexpr int kDH = 64;     // head dim

typedef __bf16 bf16_t;
typedef __attribute__((ext_vector_type(16))) __bf16 v16bf;
typedef __attribute__((ext_vector_type(8)))  float  v8f;

static __device__ __forceinline__ v8f wmma_bf16(v16bf a, v16bf b, v8f c) {
  // D = A(16x32 bf16) x B(32x16 bf16) + C(16x16 f32)
  return __builtin_amdgcn_wmma_f32_16x16x32_bf16(
      /*neg_a=*/false, a, /*neg_b=*/false, b,
      /*c_mod=*/(short)0, c, /*reuse_a=*/false, /*reuse_b=*/false);
}

// ---------------- async global->LDS (CDNA5) with safe fallback ----------
#if defined(__AMDGCN__) && __has_builtin(__builtin_amdgcn_global_load_async_to_lds_b128)
#define HAS_ASYNC_LDS 1
#else
#define HAS_ASYNC_LDS 0
#endif

#if HAS_ASYNC_LDS
typedef int v4i_t __attribute__((ext_vector_type(4)));
typedef __attribute__((address_space(1))) v4i_t* gbl_v4i_p;   // global int4*
typedef __attribute__((address_space(3))) v4i_t* lds_v4i_p;   // LDS int4*
#endif

// copy 16 bytes from global to LDS (per lane)
static __device__ __forceinline__ void async_copy_b128(const bf16_t* g,
                                                       bf16_t* l) {
#if HAS_ASYNC_LDS
  __builtin_amdgcn_global_load_async_to_lds_b128(
      (gbl_v4i_p)g, (lds_v4i_p)l, /*offset=*/0, /*cpol=*/0);
#else
  *(float4*)l = *(const float4*)g;     // through-VGPR fallback
#endif
}

static __device__ __forceinline__ void wait_async_copies() {
#if HAS_ASYNC_LDS
#if __has_builtin(__builtin_amdgcn_s_wait_asynccnt)
  __builtin_amdgcn_s_wait_asynccnt(0);
#else
  asm volatile("s_wait_asynccnt 0x0" ::: "memory");
#endif
#endif
}

// =====================================================================
// Kernel 1: sparse sense-emit.  grid = (B*S, 3 projections), block = 256.
// Writes bf16 Q/K/V in [B*H, S, 64] head-major layout.
// =====================================================================
__global__ __launch_bounds__(256) void sense_emit_kernel(
    const float* __restrict__ x,
    const float* __restrict__ qk_emb, const float* __restrict__ qk_w,
    const float* __restrict__ v_emb,  const float* __restrict__ v_w,
    const float* __restrict__ gQ, const float* __restrict__ gK,
    const float* __restrict__ gV,
    const int* __restrict__ iQ, const int* __restrict__ iK,
    const int* __restrict__ iV,
    bf16_t* __restrict__ Qws, bf16_t* __restrict__ Kws,
    bf16_t* __restrict__ Vws)
{
  const int t   = blockIdx.x;     // token 0..B*S-1
  const int p   = blockIdx.y;     // 0=Q 1=K 2=V
  const int tid = threadIdx.x;

  const float* emb  = (p == 2) ? v_emb : qk_emb;
  const float* wmat = (p == 2) ? v_w   : qk_w;
  const float* gate = (p == 0) ? gQ : (p == 1) ? gK : gV;
  const int*   idx  = (p == 0) ? iQ : (p == 1) ? iK : iV;
  bf16_t*      outp = (p == 0) ? Qws : (p == 1) ? Kws : Vws;

  const int d0 = tid * 4;
  const float4 xv = *(const float4*)(x + (size_t)t * kD + d0);

  __shared__ float red[kK][8];
  __shared__ int   rows[kK];
  __shared__ float gts[kK];
  if (tid < kK) {
    rows[tid] = idx[t * kK + tid];
    gts[tid]  = gate[t * kK + tid];
  }
  __syncthreads();

  const int lane = tid & 31;
  const int wv   = tid >> 5;

  // activations = x . emb[row]  (full-wave then cross-wave reduction)
  #pragma unroll
  for (int k = 0; k < kK; ++k) {
    const float* er = emb + (size_t)rows[k] * kD + d0;
    // pull the emit-weight row (needed in pass 2) into cache now
    __builtin_prefetch(wmat + (size_t)rows[k] * kD + d0, 0, 1);
    float pr = xv.x * er[0] + xv.y * er[1] + xv.z * er[2] + xv.w * er[3];
    pr += __shfl_xor(pr, 16, 32);
    pr += __shfl_xor(pr,  8, 32);
    pr += __shfl_xor(pr,  4, 32);
    pr += __shfl_xor(pr,  2, 32);
    pr += __shfl_xor(pr,  1, 32);
    if (lane == 0) red[k][wv] = pr;
  }
  __syncthreads();

  float acc0 = 0.f, acc1 = 0.f, acc2 = 0.f, acc3 = 0.f;
  #pragma unroll
  for (int k = 0; k < kK; ++k) {
    float act = 0.f;
    #pragma unroll
    for (int j = 0; j < 8; ++j) act += red[k][j];
    const float g = act * gts[k];
    const float* wr = wmat + (size_t)rows[k] * kD + d0;
    acc0 += g * wr[0]; acc1 += g * wr[1];
    acc2 += g * wr[2]; acc3 += g * wr[3];
  }

  // store to [b*H + head][s][dh] as bf16
  const int b = t / kS, s = t % kS;
  const int head = d0 / kDH, dh = d0 % kDH;
  bf16_t* op = outp + (((size_t)(b * kH + head)) * kS + s) * kDH + dh;
  op[0] = (bf16_t)acc0; op[1] = (bf16_t)acc1;
  op[2] = (bf16_t)acc2; op[3] = (bf16_t)acc3;
}

// =====================================================================
// Kernel 2: flash attention (causal).  grid = (S/64, B*H), block = 128.
// Each wave owns 16 query rows; 32-key K/V tiles staged in LDS (bf16)
// via GLOBAL_LOAD_ASYNC_TO_LDS_B128 (ASYNCcnt) when available.
// WMMA bf16 16x16x32: scores (2 accum x 2 d-chunks) + PV (4 accum).
// Output written bf16 in [B*S, D] (heads re-fused) for the W_O GEMM.
// =====================================================================
__global__ __launch_bounds__(128) void flash_attn_kernel(
    const bf16_t* __restrict__ Qws, const bf16_t* __restrict__ Kws,
    const bf16_t* __restrict__ Vws, bf16_t* __restrict__ Obf)
{
  const int qtile = blockIdx.x;          // 0..S/64-1
  const int bh    = blockIdx.y;          // 0..B*H-1
  const int b     = bh / kH, h = bh % kH;
  const int tid   = threadIdx.x;
  const int lane  = tid & 31;
  const int wv    = tid >> 5;
  const int half  = lane >> 4;           // 0: lanes 0-15, 1: lanes 16-31
  const int ln    = lane & 15;
  const int qbase = qtile * 64;

  __shared__ alignas(16) bf16_t Klds[32 * kDH];
  __shared__ alignas(16) bf16_t Vlds[32 * kDH];
  __shared__ alignas(16) bf16_t Qlds[64 * kDH];
  __shared__ alignas(16) bf16_t Plds[4][16 * 32];

  // ---- stage Q tile (64x64) into LDS with 1/sqrt(d_head) folded in ----
  const bf16_t* Qg = Qws + ((size_t)bh * kS + qbase) * kDH;
  for (int i = tid; i < 64 * kDH; i += 128) {
    float q = (float)Qg[i] * 0.125f;     // 1/sqrt(64)
    Qlds[i] = (bf16_t)q;
  }
  __syncthreads();

  // ---- build per-wave Q A-fragments (16x32 each, d 0-31 and 32-63) ----
  v16bf a_lo, a_hi;
  #pragma unroll
  for (int j = 0; j < 16; ++j) {
    const int dd = j + 16 * half;   // contiguous k order
    a_lo[j] = Qlds[(wv * 16 + ln) * kDH + dd];
    a_hi[j] = Qlds[(wv * 16 + ln) * kDH + 32 + dd];
  }

  v8f o0 = {}, o1 = {}, o2 = {}, o3 = {};
  float mrow[8], lrow[8];
  #pragma unroll
  for (int r = 0; r < 8; ++r) { mrow[r] = -3.0e38f; lrow[r] = 0.f; }

  const bf16_t* Kg = Kws + (size_t)bh * kS * kDH;
  const bf16_t* Vg = Vws + (size_t)bh * kS * kDH;

  const int nkt = qbase / 32 + 2;        // causal tile count (32 keys/tile)
  for (int kt = 0; kt < nkt; ++kt) {
    __syncthreads();                     // previous-tile LDS reads done
    {
      const size_t tb = (size_t)kt * (32 * kDH);   // elements
      #pragma unroll
      for (int c = 0; c < 2; ++c) {
        const int idx = tid + c * 128;   // 16-byte chunk id (256 per tile)
        async_copy_b128(Kg + tb + idx * 8, &Klds[idx * 8]);
        async_copy_b128(Vg + tb + idx * 8, &Vlds[idx * 8]);
      }
      if (kt + 1 < nkt) {                // pull next tile toward L2
        __builtin_prefetch(Kg + tb + 32 * kDH + tid * 16, 0, 1);
        __builtin_prefetch(Vg + tb + 32 * kDH + tid * 16, 0, 1);
      }
      wait_async_copies();
    }
    __syncthreads();

    // ---- scores: S0 = keys 0-15 of tile, S1 = keys 16-31 ----
    v8f S0 = {}, S1 = {};
    v16bf bk;
    #pragma unroll
    for (int j = 0; j < 16; ++j) bk[j] = Klds[(ln) * kDH + (j + 16 * half)];
    S0 = wmma_bf16(a_lo, bk, S0);
    #pragma unroll
    for (int j = 0; j < 16; ++j) bk[j] = Klds[(ln) * kDH + 32 + (j + 16 * half)];
    S0 = wmma_bf16(a_hi, bk, S0);
    #pragma unroll
    for (int j = 0; j < 16; ++j) bk[j] = Klds[(16 + ln) * kDH + (j + 16 * half)];
    S1 = wmma_bf16(a_lo, bk, S1);
    #pragma unroll
    for (int j = 0; j < 16; ++j) bk[j] = Klds[(16 + ln) * kDH + 32 + (j + 16 * half)];
    S1 = wmma_bf16(a_hi, bk, S1);

    // ---- causal mask + online softmax (per half-wave reductions) ----
    const int rowg_base = qbase + wv * 16 + 8 * half;
    const int key0 = kt * 32 + ln;
    const int key1 = kt * 32 + 16 + ln;
    float alpha[8];
    #pragma unroll
    for (int r = 0; r < 8; ++r) {
      const int rowg = rowg_base + r;
      float s0 = (key0 <= rowg) ? S0[r] : -3.0e38f;
      float s1 = (key1 <= rowg) ? S1[r] : -3.0e38f;
      float mx = fmaxf(s0, s1);
      mx = fmaxf(mx, __shfl_xor(mx, 1, 32));
      mx = fmaxf(mx, __shfl_xor(mx, 2, 32));
      mx = fmaxf(mx, __shfl_xor(mx, 4, 32));
      mx = fmaxf(mx, __shfl_xor(mx, 8, 32));
      const float mn = fmaxf(mrow[r], mx);
      alpha[r] = __expf(mrow[r] - mn);
      const float p0 = __expf(s0 - mn);
      const float p1 = __expf(s1 - mn);
      float rs = p0 + p1;
      rs += __shfl_xor(rs, 1, 32);
      rs += __shfl_xor(rs, 2, 32);
      rs += __shfl_xor(rs, 4, 32);
      rs += __shfl_xor(rs, 8, 32);
      lrow[r] = lrow[r] * alpha[r] + rs;
      mrow[r] = mn;
      const int prow = r + 8 * half;
      Plds[wv][prow * 32 + ln]      = (bf16_t)p0;
      Plds[wv][prow * 32 + 16 + ln] = (bf16_t)p1;
    }
    __builtin_amdgcn_wave_barrier();   // wave-internal LDS store->load order

    #pragma unroll
    for (int r = 0; r < 8; ++r) {
      o0[r] *= alpha[r]; o1[r] *= alpha[r];
      o2[r] *= alpha[r]; o3[r] *= alpha[r];
    }

    // ---- P (16x32 bf16 A-fragment) x V (32x64) ----
    v16bf pa;
    #pragma unroll
    for (int j = 0; j < 16; ++j) {
      const int kk = j + 16 * half;
      pa[j] = Plds[wv][ln * 32 + kk];
    }
    v16bf bv;
    #pragma unroll
    for (int j = 0; j < 16; ++j) bv[j] = Vlds[(j + 16 * half) * kDH + 0 + ln];
    o0 = wmma_bf16(pa, bv, o0);
    #pragma unroll
    for (int j = 0; j < 16; ++j) bv[j] = Vlds[(j + 16 * half) * kDH + 16 + ln];
    o1 = wmma_bf16(pa, bv, o1);
    #pragma unroll
    for (int j = 0; j < 16; ++j) bv[j] = Vlds[(j + 16 * half) * kDH + 32 + ln];
    o2 = wmma_bf16(pa, bv, o2);
    #pragma unroll
    for (int j = 0; j < 16; ++j) bv[j] = Vlds[(j + 16 * half) * kDH + 48 + ln];
    o3 = wmma_bf16(pa, bv, o3);
  }

  // ---- normalize and store (heads re-fused: d = h*64 + dh), bf16 ----
  #pragma unroll
  for (int r = 0; r < 8; ++r) {
    const float inv = 1.0f / lrow[r];
    const int srow = qbase + wv * 16 + r + 8 * half;
    const size_t base = ((size_t)(b * kS + srow)) * kD + h * kDH;
    Obf[base +  0 + ln] = (bf16_t)(o0[r] * inv);
    Obf[base + 16 + ln] = (bf16_t)(o1[r] * inv);
    Obf[base + 32 + ln] = (bf16_t)(o2[r] * inv);
    Obf[base + 48 + ln] = (bf16_t)(o3[r] * inv);
  }
}

// =====================================================================
// Kernel 3: output projection GEMM  out[4096,1024] = A x W_O  (bf16->f32).
// grid = (M/64, N/64), block = 128 (4 waves); 64x64 tile, K-chunks of 32.
// A/B tiles staged with async global->LDS copies.
// =====================================================================
__global__ __launch_bounds__(128) void gemm_wo_kernel(
    const bf16_t* __restrict__ A, const bf16_t* __restrict__ Wb,
    float* __restrict__ Out)
{
  const int m0   = blockIdx.x * 64;
  const int n0   = blockIdx.y * 64;
  const int tid  = threadIdx.x;
  const int lane = tid & 31;
  const int wv   = tid >> 5;
  const int half = lane >> 4;
  const int ln   = lane & 15;

  __shared__ alignas(16) bf16_t Alds[64 * 32];
  __shared__ alignas(16) bf16_t Blds[32 * 64];

  v8f acc0 = {}, acc1 = {}, acc2 = {}, acc3 = {};

  for (int kk = 0; kk < kD; kk += 32) {
    __syncthreads();
    #pragma unroll
    for (int c = 0; c < 2; ++c) {
      const int idx = tid + c * 128;             // 16-byte chunk id
      const int ar = idx >> 2, ac = idx & 3;     // A: 64 rows x 4 chunks/row
      async_copy_b128(A + (size_t)(m0 + ar) * kD + kk + ac * 8,
                      &Alds[ar * 32 + ac * 8]);
      const int br = idx >> 3, bc = idx & 7;     // B: 32 rows x 8 chunks/row
      async_copy_b128(Wb + (size_t)(kk + br) * kD + n0 + bc * 8,
                      &Blds[br * 64 + bc * 8]);
    }
    wait_async_copies();
    __syncthreads();

    v16bf a;
    #pragma unroll
    for (int j = 0; j < 16; ++j) {
      const int kd = j + 16 * half;
      a[j] = Alds[(wv * 16 + ln) * 32 + kd];
    }
    v16bf bf;
    #pragma unroll
    for (int j = 0; j < 16; ++j) bf[j] = Blds[(j + 16 * half) * 64 +  0 + ln];
    acc0 = wmma_bf16(a, bf, acc0);
    #pragma unroll
    for (int j = 0; j < 16; ++j) bf[j] = Blds[(j + 16 * half) * 64 + 16 + ln];
    acc1 = wmma_bf16(a, bf, acc1);
    #pragma unroll
    for (int j = 0; j < 16; ++j) bf[j] = Blds[(j + 16 * half) * 64 + 32 + ln];
    acc2 = wmma_bf16(a, bf, acc2);
    #pragma unroll
    for (int j = 0; j < 16; ++j) bf[j] = Blds[(j + 16 * half) * 64 + 48 + ln];
    acc3 = wmma_bf16(a, bf, acc3);
  }

  #pragma unroll
  for (int r = 0; r < 8; ++r) {
    const int row = m0 + wv * 16 + r + 8 * half;
    float* op = Out + (size_t)row * kD + n0;
    op[ 0 + ln] = acc0[r];
    op[16 + ln] = acc1[r];
    op[32 + ln] = acc2[r];
    op[48 + ln] = acc3[r];
  }
}

// =====================================================================
// Kernel 4: f32 -> bf16 conversion (for W_O)
// =====================================================================
__global__ __launch_bounds__(256) void cvt_bf16_kernel(
    const float* __restrict__ in, bf16_t* __restrict__ out, int n)
{
  const int i = blockIdx.x * 256 + threadIdx.x;
  if (i < n) out[i] = (bf16_t)in[i];
}

// =====================================================================
extern "C" void kernel_launch(void* const* d_in, const int* in_sizes, int n_in,
                              void* d_out, int out_size, void* d_ws,
                              size_t ws_size, hipStream_t stream)
{
  (void)in_sizes; (void)n_in; (void)out_size; (void)ws_size;

  const float* x      = (const float*)d_in[0];
  const float* qk_emb = (const float*)d_in[1];
  const float* qk_w   = (const float*)d_in[2];
  const float* v_emb  = (const float*)d_in[3];
  const float* v_w    = (const float*)d_in[4];
  const float* gQ     = (const float*)d_in[5];
  const float* gK     = (const float*)d_in[6];
  const float* gV     = (const float*)d_in[7];
  const int*   iQ     = (const int*)d_in[8];
  const int*   iK     = (const int*)d_in[9];
  const int*   iV     = (const int*)d_in[10];
  const float* W_O    = (const float*)d_in[11];
  float*       out    = (float*)d_out;

  // workspace layout (bf16): Q/K/V [B*H,S,64], attn-out [B*S,D], W_O [D,D]
  const size_t nTok = (size_t)kB * kS;          // 4096
  const size_t nQKV = nTok * kD;                // 4M elements
  char* ws = (char*)d_ws;
  bf16_t* Qws = (bf16_t*)ws;                 ws += nQKV * sizeof(bf16_t);
  bf16_t* Kws = (bf16_t*)ws;                 ws += nQKV * sizeof(bf16_t);
  bf16_t* Vws = (bf16_t*)ws;                 ws += nQKV * sizeof(bf16_t);
  bf16_t* Obf = (bf16_t*)ws;                 ws += nQKV * sizeof(bf16_t);
  bf16_t* Wbf = (bf16_t*)ws;

  // 1) sparse sense-emit -> bf16 Q/K/V in head-major layout
  dim3 g1((unsigned)nTok, 3);
  sense_emit_kernel<<<g1, 256, 0, stream>>>(
      x, qk_emb, qk_w, v_emb, v_w, gQ, gK, gV, iQ, iK, iV, Qws, Kws, Vws);

  // 2) W_O -> bf16
  cvt_bf16_kernel<<<(kD * kD + 255) / 256, 256, 0, stream>>>(W_O, Wbf, kD * kD);

  // 3) causal flash attention (bf16 WMMA, f32 accum)
  dim3 g2(kS / 64, kB * kH);
  flash_attn_kernel<<<g2, 128, 0, stream>>>(Qws, Kws, Vws, Obf);

  // 4) output projection GEMM (bf16 WMMA, f32 out)
  dim3 g3((unsigned)(nTok / 64), kD / 64);
  gemm_wo_kernel<<<g3, 128, 0, stream>>>(Obf, Wbf, out);
}
